// MultiHeadAttention_90941637525880
// MI455X (gfx1250) — compile-verified, matrix operand order checked by reading, not run on hardware
//
#include <hip/hip_runtime.h>

// Problem constants (reference: B=4, S=2048, D=1024, H=16, DH=64)
#define CB 4
#define CS 2048
#define CD 1024
#define CH 16
#define CDH 64

typedef __attribute__((ext_vector_type(16))) __bf16 v16bf;
typedef __attribute__((ext_vector_type(8)))  float  v8f;

union FragU { v16bf v; uint4 q[2]; };

__device__ __forceinline__ unsigned short f2bf(float x) {
  unsigned u = __float_as_uint(x);
  unsigned r = u + 0x7fffu + ((u >> 16) & 1u);
  return (unsigned short)(r >> 16);
}

// Load a 16-bit A/B WMMA fragment from a row-major source whose K dimension is
// contiguous.  Per CDNA5 ISA: lane = (row%16) + 16*khalf; VGPRs 0-3
// hold K = khalf*8 .. +7 (packed pairs), VGPRs 4-7 hold K = 16+khalf*8 .. +7.
__device__ __forceinline__ v16bf load_frag_g(const unsigned short* __restrict__ p, int off) {
  FragU f;
  f.q[0] = *reinterpret_cast<const uint4*>(p + off);
  f.q[1] = *reinterpret_cast<const uint4*>(p + off + 16);
  return f.v;
}

// Async 16-byte global -> LDS copy (CDNA5 GLOBAL_LOAD_ASYNC_TO_LDS_B128,
// tracked by ASYNCcnt).  Generic pointers to __shared__ carry the LDS byte
// offset in their low 32 bits (aperture scheme), so pass that as VDST.
__device__ __forceinline__ void async_cp16(const unsigned short* g, unsigned short* l) {
  unsigned lds_addr = (unsigned)(unsigned long long)(uintptr_t)l;
  unsigned long long gaddr = (unsigned long long)(uintptr_t)g;
  asm volatile("global_load_async_to_lds_b128 %0, %1, off"
               :: "v"(lds_addr), "v"(gaddr) : "memory");
}
__device__ __forceinline__ void wait_async0() {
  asm volatile("s_wait_asynccnt 0x0" ::: "memory");
}

__device__ __forceinline__ float hmax16(float v) {
  v = fmaxf(v, __shfl_xor(v, 1, 16));
  v = fmaxf(v, __shfl_xor(v, 2, 16));
  v = fmaxf(v, __shfl_xor(v, 4, 16));
  v = fmaxf(v, __shfl_xor(v, 8, 16));
  return v;
}
__device__ __forceinline__ float hsum16(float v) {
  v += __shfl_xor(v, 1, 16);
  v += __shfl_xor(v, 2, 16);
  v += __shfl_xor(v, 4, 16);
  v += __shfl_xor(v, 8, 16);
  return v;
}

// ---------------- conversion kernels ----------------

__global__ void cvt_x_bf16(const float* __restrict__ x, unsigned short* __restrict__ xb, int n) {
  int i = blockIdx.x * blockDim.x + threadIdx.x;
  if (i < n) xb[i] = f2bf(x[i]);
}

// Wt[sel][h][e][k] = W_sel[h][k][e]  (K contiguous -> B-operand ready)
__global__ void cvt_wqkv(const float* __restrict__ Wq, const float* __restrict__ Wk,
                         const float* __restrict__ Wv, unsigned short* __restrict__ Wt) {
  int i = blockIdx.x * blockDim.x + threadIdx.x;
  if (i >= 3 * CH * CDH * CD) return;
  int k = i & (CD - 1);
  int r = i >> 10;
  int e = r & (CDH - 1); r >>= 6;
  int h = r & (CH - 1);
  int sel = r >> 4;
  const float* W = (sel == 0) ? Wq : (sel == 1) ? Wk : Wv;
  Wt[i] = f2bf(W[((size_t)h * CD + k) * CDH + e]);
}

// Wot[n][k] = Wo[k][n]
__global__ void cvt_wo(const float* __restrict__ Wo, unsigned short* __restrict__ Wot) {
  int i = blockIdx.x * blockDim.x + threadIdx.x;
  if (i >= CD * CD) return;
  int k = i & (CD - 1);
  int n = i >> 10;
  Wot[i] = f2bf(Wo[(size_t)k * CD + n]);
}

// ---------------- QKV projection ----------------
// grid = (S/128, B*H, 3), block = 256 (8 waves, one 16x64 tile per wave)
__global__ __launch_bounds__(256) void qkv_proj(
    const unsigned short* __restrict__ xb, const unsigned short* __restrict__ Wt,
    const float* __restrict__ bq, const float* __restrict__ bk, const float* __restrict__ bv,
    unsigned short* __restrict__ Qb, unsigned short* __restrict__ Kb,
    unsigned short* __restrict__ Vtb) {
  const int lane = threadIdx.x & 31, wave = threadIdx.x >> 5;
  const int sel = blockIdx.z;
  const int bh = blockIdx.y;
  const int b = bh >> 4, h = bh & 15;
  const int m0 = (blockIdx.x * 8 + wave) * 16;
  const int mrow = lane & 15, khalf = lane >> 4;

  const unsigned short* A = xb + (size_t)b * CS * CD;
  const unsigned short* Bw = Wt + ((size_t)sel * CH + h) * CDH * CD;

  v8f acc[4] = {};
  for (int k0 = 0; k0 < CD; k0 += 32) {
    v16bf a = load_frag_g(A, (m0 + mrow) * CD + k0 + khalf * 8);
#pragma unroll
    for (int t = 0; t < 4; ++t) {
      v16bf bf = load_frag_g(Bw, (t * 16 + mrow) * CD + k0 + khalf * 8);
      acc[t] = __builtin_amdgcn_wmma_f32_16x16x32_bf16(false, a, false, bf,
                                                       (short)0, acc[t], false, false);
    }
  }

  const float* bias = (sel == 0) ? bq : (sel == 1) ? bk : bv;
  if (sel < 2) {
    unsigned short* Out = ((sel == 0) ? Qb : Kb) + (size_t)bh * CS * CDH;
    const float qscale = (sel == 0) ? 0.125f : 1.0f;  // fold 1/sqrt(DH)=2^-3 into Q
#pragma unroll
    for (int t = 0; t < 4; ++t) {
      int e = t * 16 + mrow;
      float bb = bias[h * CDH + e];
#pragma unroll
      for (int r = 0; r < 8; ++r) {
        int s = m0 + r + 8 * khalf;
        Out[(size_t)s * CDH + e] = f2bf((acc[t][r] + bb) * qscale);
      }
    }
  } else {
    // V stored transposed: Vt[bh][e][t], t contiguous -> b128-packable rows
    unsigned short* Out = Vtb + (size_t)bh * CDH * CS;
#pragma unroll
    for (int t = 0; t < 4; ++t) {
      int e = t * 16 + mrow;
      float bb = bias[h * CDH + e];
      union { unsigned short s[8]; uint4 q; } pk;
#pragma unroll
      for (int r = 0; r < 8; ++r) pk.s[r] = f2bf(acc[t][r] + bb);
      *reinterpret_cast<uint4*>(Out + (size_t)e * CS + m0 + 8 * khalf) = pk.q;
    }
  }
}

// ---------------- flash attention ----------------
// Block = 8 waves, all on the same (b,h); waves cover 8 consecutive 16-row
// query tiles.  K/V 32-key blocks are staged into LDS with async global->LDS
// copies (ASYNCcnt), double buffered: issue next block, compute current,
// wait own async, barrier.
__global__ __launch_bounds__(256) void attn_fwd(
    const unsigned short* __restrict__ Qb, const unsigned short* __restrict__ Kb,
    const unsigned short* __restrict__ Vtb, unsigned short* __restrict__ Cb) {
  __shared__ __align__(16) unsigned short ksh[2][32][CDH];  // [key][e]  8 KB
  __shared__ __align__(16) unsigned short vsh[2][CDH][32];  // [e][key]  8 KB
  __shared__ __align__(16) unsigned short pbuf[8][16 * 32]; // per-wave P  8 KB

  const int tid = threadIdx.x;
  const int lane = tid & 31, wave = tid >> 5;
  const int bh = blockIdx.x >> 4;            // 16 blocks per (b,h)
  const int mt = (blockIdx.x & 15) * 8 + wave;
  const int m0 = mt * 16;
  const int mrow = lane & 15, khalf = lane >> 4;
  const int b = bh >> 4, h = bh & 15;

  const unsigned short* Q = Qb + (size_t)bh * CS * CDH;
  const unsigned short* K = Kb + (size_t)bh * CS * CDH;
  const unsigned short* V = Vtb + (size_t)bh * CDH * CS;

  // cooperative staging indices: 256 threads x 16B cover each 4 KB block
  const int skr = tid >> 3, skc = (tid & 7) * 8;  // K: 32 rows x 64 e
  const int svr = tid >> 2, svc = (tid & 3) * 8;  // V: 64 rows x 32 t

  // Q fragments for the whole row tile (DH=64 -> two K=32 fragments)
  v16bf qf0 = load_frag_g(Q, (m0 + mrow) * CDH + 0 + khalf * 8);
  v16bf qf1 = load_frag_g(Q, (m0 + mrow) * CDH + 32 + khalf * 8);

  v8f o[4] = {};
  float mi[8], li[8];
#pragma unroll
  for (int r = 0; r < 8; ++r) { mi[r] = -1.0e30f; li[r] = 0.0f; }

  const float LOG2E = 1.4426950408889634f;

  // prologue: async-stage first key block into buffer 0
  async_cp16(K + (size_t)skr * CDH + skc, &ksh[0][skr][skc]);
  async_cp16(V + (size_t)svr * CS + svc, &vsh[0][svr][svc]);
  wait_async0();
  __syncthreads();

  int buf = 0;
  for (int it = 0; it < CS / 32; ++it) {
    const int t0 = it * 32;

    if (it + 1 < CS / 32) {  // issue async stage of next block into other buffer
      async_cp16(K + (size_t)(t0 + 32 + skr) * CDH + skc, &ksh[buf ^ 1][skr][skc]);
      async_cp16(V + (size_t)svr * CS + t0 + 32 + svc, &vsh[buf ^ 1][svr][svc]);
    }

    // scores: S[16x32] = Q(16x64) @ K_block^T(64x32), K from LDS
    v8f sc[2] = {};
#pragma unroll
    for (int nt = 0; nt < 2; ++nt) {
#pragma unroll
      for (int kc = 0; kc < 2; ++kc) {
        FragU kf;
        kf.q[0] = *reinterpret_cast<const uint4*>(&ksh[buf][nt * 16 + mrow][kc * 32 + khalf * 8]);
        kf.q[1] = *reinterpret_cast<const uint4*>(&ksh[buf][nt * 16 + mrow][kc * 32 + 16 + khalf * 8]);
        v16bf qf = (kc == 0) ? qf0 : qf1;
        sc[nt] = __builtin_amdgcn_wmma_f32_16x16x32_bf16(false, qf, false, kf.v,
                                                         (short)0, sc[nt], false, false);
      }
    }

    // online softmax (rows live across lanes 0-15 / 16-31 of the half-wave)
    float f[8];
#pragma unroll
    for (int r = 0; r < 8; ++r) {
      float sm = hmax16(fmaxf(sc[0][r], sc[1][r]));
      float mnew = fmaxf(mi[r], sm);
      f[r] = exp2f((mi[r] - mnew) * LOG2E);
      float p0 = exp2f((sc[0][r] - mnew) * LOG2E);
      float p1 = exp2f((sc[1][r] - mnew) * LOG2E);
      li[r] = li[r] * f[r] + hsum16(p0 + p1);
      mi[r] = mnew;
      int row = r + 8 * khalf;
      pbuf[wave][row * 32 + mrow] = f2bf(p0);
      pbuf[wave][row * 32 + 16 + mrow] = f2bf(p1);
    }
#pragma unroll
    for (int t = 0; t < 4; ++t)
#pragma unroll
      for (int r = 0; r < 8; ++r) o[t][r] *= f[r];

    // same-wave DS ordering: P stores above must land before fragment reloads
    asm volatile("s_wait_dscnt 0x0" ::: "memory");

    FragU pf;
    pf.q[0] = *reinterpret_cast<const uint4*>(&pbuf[wave][mrow * 32 + khalf * 8]);
    pf.q[1] = *reinterpret_cast<const uint4*>(&pbuf[wave][mrow * 32 + 16 + khalf * 8]);

    // O += P(16x32) @ V_block(32x64), V from LDS (key-contiguous rows)
#pragma unroll
    for (int t = 0; t < 4; ++t) {
      FragU vf;
      vf.q[0] = *reinterpret_cast<const uint4*>(&vsh[buf][t * 16 + mrow][khalf * 8]);
      vf.q[1] = *reinterpret_cast<const uint4*>(&vsh[buf][t * 16 + mrow][16 + khalf * 8]);
      o[t] = __builtin_amdgcn_wmma_f32_16x16x32_bf16(false, pf.v, false, vf.v,
                                                     (short)0, o[t], false, false);
    }

    // our async copies into buf^1 must have landed; then block-wide rendezvous
    wait_async0();
    __syncthreads();
    buf ^= 1;
  }

  // epilogue: normalize and write concat[b, s, h*64+e] as bf16
  float inv[8];
#pragma unroll
  for (int r = 0; r < 8; ++r) inv[r] = 1.0f / li[r];
#pragma unroll
  for (int t = 0; t < 4; ++t) {
    int e = h * CDH + t * 16 + mrow;
#pragma unroll
    for (int r = 0; r < 8; ++r) {
      int s = m0 + r + 8 * khalf;
      Cb[((size_t)b * CS + s) * CD + e] = f2bf(o[t][r] * inv[r]);
    }
  }
}

// ---------------- output projection ----------------
__global__ __launch_bounds__(256) void out_proj(
    const unsigned short* __restrict__ Cb, const unsigned short* __restrict__ Wot,
    const float* __restrict__ bo, float* __restrict__ out) {
  const int lane = threadIdx.x & 31, wave = threadIdx.x >> 5;
  const int id = blockIdx.x * 8 + wave;
  const int rt = id >> 4;          // row tile 0..511
  const int n0 = (id & 15) * 64;   // output-column block
  const int m0 = rt * 16;
  const int mrow = lane & 15, khalf = lane >> 4;

  v8f acc[4] = {};
  for (int k0 = 0; k0 < CD; k0 += 32) {
    v16bf a = load_frag_g(Cb, (m0 + mrow) * CD + k0 + khalf * 8);
#pragma unroll
    for (int t = 0; t < 4; ++t) {
      v16bf bf = load_frag_g(Wot, (n0 + t * 16 + mrow) * CD + k0 + khalf * 8);
      acc[t] = __builtin_amdgcn_wmma_f32_16x16x32_bf16(false, a, false, bf,
                                                       (short)0, acc[t], false, false);
    }
  }
#pragma unroll
  for (int t = 0; t < 4; ++t) {
    int col = n0 + t * 16 + mrow;
    float bb = bo[col];
#pragma unroll
    for (int r = 0; r < 8; ++r) {
      int s = m0 + r + 8 * khalf;
      out[(size_t)s * CD + col] = acc[t][r] + bb;
    }
  }
}

// ---------------- launch ----------------
extern "C" void kernel_launch(void* const* d_in, const int* in_sizes, int n_in,
                              void* d_out, int out_size, void* d_ws, size_t ws_size,
                              hipStream_t stream) {
  (void)in_sizes; (void)n_in; (void)out_size; (void)ws_size;
  const float* x  = (const float*)d_in[0];
  const float* Wq = (const float*)d_in[1];
  const float* bq = (const float*)d_in[2];
  const float* Wk = (const float*)d_in[3];
  const float* bk = (const float*)d_in[4];
  const float* Wv = (const float*)d_in[5];
  const float* bv = (const float*)d_in[6];
  const float* Wo = (const float*)d_in[7];
  const float* bo = (const float*)d_in[8];
  float* out = (float*)d_out;

  char* ws = (char*)d_ws;
  size_t off = 0;
  unsigned short* xb  = (unsigned short*)(ws + off); off += (size_t)CB * CS * CD * 2;       // 16 MB
  unsigned short* Wt  = (unsigned short*)(ws + off); off += (size_t)3 * CH * CDH * CD * 2;  // 6 MB
  unsigned short* Wot = (unsigned short*)(ws + off); off += (size_t)CD * CD * 2;            // 2 MB
  unsigned short* Qb  = (unsigned short*)(ws + off); off += (size_t)CB * CH * CS * CDH * 2; // 16 MB
  unsigned short* Kb  = (unsigned short*)(ws + off); off += (size_t)CB * CH * CS * CDH * 2; // 16 MB
  unsigned short* Vtb = (unsigned short*)(ws + off); off += (size_t)CB * CH * CDH * CS * 2; // 16 MB
  unsigned short* Cb  = (unsigned short*)(ws + off); off += (size_t)CB * CS * CD * 2;       // 16 MB

  const int nx = CB * CS * CD;
  cvt_x_bf16<<<(nx + 255) / 256, 256, 0, stream>>>(x, xb, nx);
  cvt_wqkv<<<(3 * CH * CDH * CD + 255) / 256, 256, 0, stream>>>(Wq, Wk, Wv, Wt);
  cvt_wo<<<(CD * CD + 255) / 256, 256, 0, stream>>>(Wo, Wot);

  qkv_proj<<<dim3(CS / 128, CB * CH, 3), 256, 0, stream>>>(xb, Wt, bq, bk, bv, Qb, Kb, Vtb);

  attn_fwd<<<CB * CH * 16, 256, 0, stream>>>(Qb, Kb, Vtb, Cb);

  out_proj<<<((CB * CS / 16) * (CD / 64)) / 8, 256, 0, stream>>>(Cb, Wot, bo, out);
}
